// MultiHeadAttention_22196390986016
// MI455X (gfx1250) — compile-verified, matrix operand order checked by reading, not run on hardware
//
#include <hip/hip_runtime.h>
#include <hip/hip_bf16.h>

// ---------------------------------------------------------------------------
// MHA forward for gfx1250 (MI455X). bf16 WMMA everywhere, f32 accumulate.
// B=2, S=2048, D=1024, H=16, HD=64.
// Attention computes S^T = K*Q^T so softmax stats are per-lane scalars and
// the P fragment feeds the PV WMMA directly from registers (no LDS).
// Causal mask is applied only on the single diagonal tile per wave.
// ---------------------------------------------------------------------------

#define B_  2
#define S_  2048
#define D_  1024
#define H_  16
#define HD_ 64
#define M_  (B_ * S_)   // 4096 rows for the projections

typedef __attribute__((ext_vector_type(16))) __bf16        v16bf;
typedef __attribute__((ext_vector_type(8)))  float         v8f;
typedef __attribute__((ext_vector_type(4)))  unsigned int  v4u;

union FragU { v16bf v; v4u u[2]; };

// Per-lane 16-element bf16 fragment: elements 0..7 from base+klo, 8..15 from
// base+klo+16 (documented 16-bit WMMA per-lane K-chunk layout, two b128 loads).
__device__ __forceinline__ v16bf load_frag(const __bf16* base, int klo) {
    FragU f;
    f.u[0] = *(const v4u*)(base + klo);
    f.u[1] = *(const v4u*)(base + klo + 16);
    return f.v;
}

__device__ __forceinline__ v8f wmma_bf16(v16bf a, v16bf b, v8f c) {
    return __builtin_amdgcn_wmma_f32_16x16x32_bf16(
        /*neg_a=*/false, a, /*neg_b=*/false, b,
        /*c_mod=*/(short)0, c, /*reuse_a=*/false, /*reuse_b=*/false);
}

// ---------------------------------------------------------------------------
// f32 -> bf16 conversion, 4 elements per thread.
// ---------------------------------------------------------------------------
__global__ void cvt_bf16_v4(const float* __restrict__ in,
                            __bf16* __restrict__ out, int n4) {
    int i = blockIdx.x * blockDim.x + threadIdx.x;
    if (i < n4) {
        float4 f = ((const float4*)in)[i];
        union { __bf16 h[4]; unsigned long long u; } o;
        o.h[0] = (__bf16)f.x; o.h[1] = (__bf16)f.y;
        o.h[2] = (__bf16)f.z; o.h[3] = (__bf16)f.w;
        *(unsigned long long*)(out + (size_t)i * 4) = o.u;
    }
}

// ---------------------------------------------------------------------------
// Tiled WMMA GEMM:  Y[m,n] = sum_k A[m,k] * W[n,k] + bias[n]
//   A: [M,K] bf16 row-major,  W: [N,K] bf16 row-major (nn.Linear weight)
// MODE 0: store bf16 row-major [M,N]
// MODE 1: store bf16 head-transposed V^T: out[((b*H + h)*64 + d)*S + s]
// MODE 2: store f32 row-major [M,N]
// Block: 256 threads = 8 waves. Tile 128(M) x 64(N). K-step 32.
// ---------------------------------------------------------------------------
template <int MODE>
__global__ void gemm_wmma(const __bf16* __restrict__ A,
                          const __bf16* __restrict__ W,
                          const float*  __restrict__ bias,
                          void* __restrict__ out,
                          int M, int N, int K) {
    __shared__ __align__(16) __bf16 ldsB[64 * 32];

    const int tid = threadIdx.x;
    const int w   = tid >> 5;          // wave id 0..7
    const int l   = tid & 31;          // lane
    const int lan = l & 15;
    const int klo = (l >> 4) * 8;

    const int mblock = blockIdx.y * 128;
    const int nblock = blockIdx.x * 64;
    const int mbase  = mblock + w * 16;

    const __bf16* arow = A + (size_t)(mbase + lan) * K;

    v8f acc[4] = {};

    for (int kb = 0; kb < K; kb += 32) {
        __syncthreads();
        // Cooperative B tile load: 64 rows x 32 K values (4 KB), 16 B/thread.
        {
            int n  = tid >> 2;           // 0..63
            int kc = (tid & 3) * 8;      // 0,8,16,24
            *(v4u*)&ldsB[n * 32 + kc] =
                *(const v4u*)&W[(size_t)(nblock + n) * K + kb + kc];
        }
        __syncthreads();

        if (kb + 64 < K) __builtin_prefetch(arow + kb + 64, 0, 1);

        v16bf af = load_frag(arow + kb, klo);
#pragma unroll
        for (int nt = 0; nt < 4; ++nt) {
            v16bf bf = load_frag(&ldsB[(nt * 16 + lan) * 32], klo);
            acc[nt] = wmma_bf16(af, bf, acc[nt]);
        }
    }

    // Store with bias. C layout: lane holds column n (lan), VGPR i holds row
    // mbase + i + 8*(l>=16).
    const int rowoff = (l >> 4) * 8;
#pragma unroll
    for (int nt = 0; nt < 4; ++nt) {
        int n   = nblock + nt * 16 + lan;
        float bv = bias[n];
#pragma unroll
        for (int i = 0; i < 8; ++i) {
            int m = mbase + i + rowoff;
            float val = acc[nt][i] + bv;
            if (MODE == 0) {
                ((__bf16*)out)[(size_t)m * N + n] = (__bf16)val;
            } else if (MODE == 2) {
                ((float*)out)[(size_t)m * N + n] = val;
            } else {
                int bb = m / S_, s = m % S_;
                int h = n >> 6, d = n & 63;
                ((__bf16*)out)[(((size_t)bb * H_ + h) * 64 + d) * S_ + s] =
                    (__bf16)val;
            }
        }
    }
}

// ---------------------------------------------------------------------------
// One 32-key attention tile. MASKED=true only for the diagonal tile.
// Lane l owns query column q = q0 + (l&15); partner lane (l^16) holds the
// other 16 keys of the same query.
// ---------------------------------------------------------------------------
template <bool MASKED>
__device__ __forceinline__ void attn_tile(
        int kb, int q, int lan, int klo,
        const __bf16* __restrict__ kbase, const __bf16* __restrict__ vbase,
        v16bf qf0, v16bf qf1, v8f (&oacc)[4], float& rowm, float& rowl) {
    const float SCALE = 0.125f;        // 1/sqrt(64)
    const float NEG   = -1e30f;

    // ---- S^T tile: A = K rows (16 keys x 64 d), B = Q^T ----
    v8f sacc[2] = {};
#pragma unroll
    for (int nt = 0; nt < 2; ++nt) {
        const __bf16* krow = kbase + (size_t)(kb + nt * 16 + lan) * D_;
        v16bf kfa = load_frag(krow,      klo);
        v16bf kfb = load_frag(krow + 32, klo);
        sacc[nt] = wmma_bf16(kfa, qf0, sacc[nt]);
        sacc[nt] = wmma_bf16(kfb, qf1, sacc[nt]);
    }

    // ---- per-lane scale (+ causal mask on the diagonal tile only) ----
    float s[16];
#pragma unroll
    for (int nt = 0; nt < 2; ++nt)
#pragma unroll
        for (int i = 0; i < 8; ++i) {
            float v = sacc[nt][i] * SCALE;
            if (MASKED) {
                int key = kb + nt * 16 + i + klo;   // klo == row offset 0|8
                if (key > q) v = NEG;
            }
            s[nt * 8 + i] = v;
        }

    // ---- online softmax: 15 in-lane max + 1 cross-half shfl ----
    float tm = s[0];
#pragma unroll
    for (int j = 1; j < 16; ++j) tm = fmaxf(tm, s[j]);
    tm = fmaxf(tm, __shfl_xor(tm, 16, 32));
    float mnew = fmaxf(rowm, tm);

    union { v16bf v; __bf16 h16[16]; } pf;
    float ps = 0.0f;
#pragma unroll
    for (int j = 0; j < 16; ++j) {
        float p = __expf(s[j] - mnew);
        ps += p;
        pf.h16[j] = (__bf16)p;
    }
    ps += __shfl_xor(ps, 16, 32);

    float corr = __expf(rowm - mnew);
    rowm = mnew;
    rowl = rowl * corr + ps;

    // ---- O^T += V * P : A = V^T rows (16 d x 32 keys), B = P ----
#pragma unroll
    for (int dt = 0; dt < 4; ++dt) {
#pragma unroll
        for (int i = 0; i < 8; ++i) oacc[dt][i] *= corr;
        v16bf vf = load_frag(vbase + (size_t)(dt * 16 + lan) * S_ + kb, klo);
        oacc[dt] = wmma_bf16(vf, pf.v, oacc[dt]);
    }
}

// ---------------------------------------------------------------------------
// Flash attention (causal), one wave per 16 query rows of one (b,h).
// Qb, Kb: bf16 [B,S,D]; Vt: bf16 [B,H,64,S]; out: bf16 [B,S,D].
// Block: 256 threads = 8 independent waves. No LDS.
// ---------------------------------------------------------------------------
__global__ void attn_wmma(const __bf16* __restrict__ Qb,
                          const __bf16* __restrict__ Kb,
                          const __bf16* __restrict__ Vt,
                          __bf16* __restrict__ outb) {
    const int w   = threadIdx.x >> 5;
    const int l   = threadIdx.x & 31;
    const int lan = l & 15;
    const int klo = (l >> 4) * 8;       // K-chunk base; also M-row offset (0|8)

    const int wid  = blockIdx.x * 8 + w;               // 0..4095
    const int qblk = wid & (S_ / 16 - 1);              // 0..127
    const int rest = wid >> 7;
    const int h    = rest & (H_ - 1);
    const int b    = rest >> 4;
    const int q0   = qblk * 16;
    const int q    = q0 + lan;                         // this lane's query row

    // Q^T B-fragments: column q = lan, chunks along head_dim (two 32-d slabs).
    const __bf16* qrow = Qb + ((size_t)(b * S_ + q) * D_ + h * 64);
    v16bf qf0 = load_frag(qrow,      klo);
    v16bf qf1 = load_frag(qrow + 32, klo);

    const __bf16* kbase = Kb + ((size_t)(b * S_) * D_ + h * 64);
    const __bf16* vbase = Vt + ((size_t)(b * H_ + h) * 64) * S_;

    v8f oacc[4] = {};                  // O^T: lane = column q, rows = d
    float rowm = -1e30f, rowl = 0.0f;

    // Tile kt is partial iff kt*32+31 > q0; with nkt = (q0+47)>>5 exactly the
    // last tile is partial and all earlier tiles are full (q0 multiple of 16).
    const int nkt = (q0 + 47) >> 5;
    for (int kt = 0; kt < nkt - 1; ++kt)
        attn_tile<false>(kt * 32, q, lan, klo, kbase, vbase,
                         qf0, qf1, oacc, rowm, rowl);
    attn_tile<true>((nkt - 1) * 32, q, lan, klo, kbase, vbase,
                    qf0, qf1, oacc, rowm, rowl);

    // ---- normalize + packed b128 stores: lane writes its query row ----
    const float inv = 1.0f / rowl;
    __bf16* orow = outb + (size_t)(b * S_ + q) * D_ + h * 64;
#pragma unroll
    for (int dt = 0; dt < 4; ++dt) {
        union { __bf16 h8[8]; v4u u; } o;
#pragma unroll
        for (int i = 0; i < 8; ++i) o.h8[i] = (__bf16)(oacc[dt][i] * inv);
        *(v4u*)(orow + dt * 16 + klo) = o.u;   // d = dt*16 + klo + 0..7
    }
}

// ---------------------------------------------------------------------------
// Launch
// ---------------------------------------------------------------------------
extern "C" void kernel_launch(void* const* d_in, const int* in_sizes, int n_in,
                              void* d_out, int out_size, void* d_ws, size_t ws_size,
                              hipStream_t stream) {
    const float* x  = (const float*)d_in[0];
    // d_in[1] = mask (causal; computed analytically in-kernel)
    const float* Wq = (const float*)d_in[2];
    const float* bq = (const float*)d_in[3];
    const float* Wk = (const float*)d_in[4];
    const float* bk = (const float*)d_in[5];
    const float* Wv = (const float*)d_in[6];
    const float* bv = (const float*)d_in[7];
    const float* Wo = (const float*)d_in[8];
    const float* bo = (const float*)d_in[9];

    char* ws = (char*)d_ws;
    const size_t MB = 1ull << 20;
    __bf16* xb  = (__bf16*)(ws);            //  8 MB  x  bf16 [B,S,D]
    __bf16* wqb = (__bf16*)(ws +  8 * MB);  //  2 MB
    __bf16* wkb = (__bf16*)(ws + 10 * MB);  //  2 MB
    __bf16* wvb = (__bf16*)(ws + 12 * MB);  //  2 MB
    __bf16* wob = (__bf16*)(ws + 14 * MB);  //  2 MB
    __bf16* Qb  = (__bf16*)(ws + 16 * MB);  //  8 MB  [B,S,D]
    __bf16* Kb  = (__bf16*)(ws + 24 * MB);  //  8 MB  [B,S,D]
    __bf16* Vt  = (__bf16*)(ws + 32 * MB);  //  8 MB  [B,H,64,S]
    __bf16* Ab  = (__bf16*)(ws + 40 * MB);  //  8 MB  attn out [B,S,D]

    // f32 -> bf16 conversions
    {
        int n4 = (B_ * S_ * D_) / 4;
        cvt_bf16_v4<<<(n4 + 255) / 256, 256, 0, stream>>>(x, xb, n4);
        int w4 = (D_ * D_) / 4;
        cvt_bf16_v4<<<(w4 + 255) / 256, 256, 0, stream>>>(Wq, wqb, w4);
        cvt_bf16_v4<<<(w4 + 255) / 256, 256, 0, stream>>>(Wk, wkb, w4);
        cvt_bf16_v4<<<(w4 + 255) / 256, 256, 0, stream>>>(Wv, wvb, w4);
        cvt_bf16_v4<<<(w4 + 255) / 256, 256, 0, stream>>>(Wo, wob, w4);
    }

    dim3 ggrid(D_ / 64, M_ / 128);   // (16, 32)
    gemm_wmma<0><<<ggrid, 256, 0, stream>>>(xb, wqb, bq, Qb, M_, D_, D_);
    gemm_wmma<0><<<ggrid, 256, 0, stream>>>(xb, wkb, bk, Kb, M_, D_, D_);
    gemm_wmma<1><<<ggrid, 256, 0, stream>>>(xb, wvb, bv, Vt, M_, D_, D_);

    // 4096 q-tiles of 16 rows, 8 waves per block
    attn_wmma<<<(B_ * H_ * (S_ / 16)) / 8, 256, 0, stream>>>(Qb, Kb, Vt, Ab);

    gemm_wmma<2><<<ggrid, 256, 0, stream>>>(Ab, wob, bo, d_out, M_, D_, D_);
}
